// CrossAttention_LR_65249143160950
// MI455X (gfx1250) — matrix operand order repaired and checked by reading, not compiled, not run on hardware
//
#include <hip/hip_runtime.h>
#include <hip/hip_bf16.h>

// ---------------- problem constants ----------------
#define BB      16
#define CC      512          // channels
#define NN      1024         // W*H tokens
#define MM      77           // context tokens
#define CTXD    768          // context dim
#define NHEADS  8
#define DH      64
#define INNER   512
#define CTXP    96           // padded ctx keys: 1 null + 77 + 18 zero-pad (mult of 32)
#define EPSV    1e-5f
#define PLD     104          // LDS row stride for P staging (>=96, padded)

typedef __attribute__((ext_vector_type(16))) __bf16 v16bf;
typedef __attribute__((ext_vector_type(8)))  float  v8f;
typedef __attribute__((ext_vector_type(4)))  unsigned int u32x4;
typedef __attribute__((ext_vector_type(8)))  int i32x8;
typedef __attribute__((ext_vector_type(4)))  int i32x4;

static __device__ inline v8f wmma_bf16(v16bf a, v16bf b, v8f c) {
    return __builtin_amdgcn_wmma_f32_16x16x32_bf16(
        false, a, false, b, (short)0, c, false, false);
}

// ---------------- Tensor Data Mover: 1-D contiguous tile -> LDS ----------------
// D# per CDNA5 ISA: group0 {count=1, lds_addr, global_addr,
// type=2}; group1 {data_size=8B, tensor_dim0=2^30 (no OOB), tile_dim0=nbytes/8}.
static __device__ inline void tdm_load_1d(const void* gsrc, void* ldst, unsigned nbytes) {
#if __has_builtin(__builtin_amdgcn_tensor_load_to_lds)
    unsigned lds = (unsigned)(unsigned long long)ldst;   // low 32 bits of flat = LDS offset
    unsigned long long ga = (unsigned long long)gsrc;
    unsigned n8 = nbytes >> 3;                           // 8-byte units
    u32x4 g0;
    g0[0] = 1u;                                          // count=1, user descriptor
    g0[1] = lds;                                         // lds_addr
    g0[2] = (unsigned)ga;                                // global_addr[31:0]
    g0[3] = (unsigned)((ga >> 32) & 0x01ffffffu) | (2u << 30);  // addr[56:32], type=2
    i32x8 g1;
    g1[0] = (int)(3u << 16);       // workgroup_mask=0, data_size=8B, no flags
    g1[1] = 0;                     // barrier addr=0; tensor_dim0[15:0]=0 (dim0 = 1<<30)
    g1[2] = (int)(0x4000u | (1u << 16)); // tensor_dim0[31:16]=0x4000, tensor_dim1=1 (lo)
    g1[3] = (int)(n8 << 16);       // tensor_dim1 hi=0, tile_dim0 = n8
    g1[4] = 1;                     // tile_dim1=1, tile_dim2=0
    g1[5] = 0x40000000;            // tensor_dim0_stride lo32 = 1<<30
    g1[6] = 0;                     // stride hi16 = 0, dim1_stride lo16 = 0
    g1[7] = 0x4000;                // tensor_dim1_stride[47:16] -> 1<<30
    i32x4 gz = {0, 0, 0, 0};
#if __clang_major__ >= 23
    i32x8 gz8 = {};
    __builtin_amdgcn_tensor_load_to_lds(g0, g1, gz, gz, gz8, 0);
#else
    __builtin_amdgcn_tensor_load_to_lds(g0, g1, gz, gz, 0);
#endif
#else
    // fallback: wave-cooperative copy (caller gates to one wave)
    const uint4* s = (const uint4*)gsrc;
    uint4* d = (uint4*)ldst;
    for (unsigned i = (threadIdx.x & 31); i < (nbytes >> 4); i += 32) d[i] = s[i];
#endif
}
static __device__ inline void tdm_wait() {
#if __has_builtin(__builtin_amdgcn_s_wait_tensorcnt)
    __builtin_amdgcn_s_wait_tensorcnt(0);
#endif
}

// ---- fragment loaders (wave32), contiguous k order k = 16*(lane/16) + element ----
// A 16x32 bf16, row-major source, leading dim ld.
static __device__ inline v16bf load_a_frag(const __bf16* src, int ld) {
    int lane = threadIdx.x & 31;
    int m    = lane & 15;
    int half = lane >> 4;
    v16bf a;
#pragma unroll
    for (int r = 0; r < 8; ++r) {
        int k0 = half * 16 + r * 2;
        a[2*r]   = src[(size_t)m * ld + k0];
        a[2*r+1] = src[(size_t)m * ld + k0 + 1];
    }
    return a;
}

// B 32x16 bf16 from row-major KxN source.
static __device__ inline v16bf load_b_frag(const __bf16* src, int ld) {
    int lane  = threadIdx.x & 31;
    int n     = lane & 15;
    int khalf = lane >> 4;
    v16bf b;
#pragma unroll
    for (int r = 0; r < 8; ++r) {
        int k0 = khalf * 16 + r * 2;
        b[2*r]   = src[(size_t)k0 * ld + n];
        b[2*r+1] = src[(size_t)(k0 + 1) * ld + n];
    }
    return b;
}

// B 32x16 = transpose of row-major NxK source (Q@K^T; K stored [keys][d]).
static __device__ inline v16bf load_bt_frag(const __bf16* src, int ld) {
    int lane  = threadIdx.x & 31;
    int n     = lane & 15;
    int khalf = lane >> 4;
    v16bf b;
#pragma unroll
    for (int r = 0; r < 8; ++r) {
        int k0 = khalf * 16 + r * 2;
        b[2*r]   = src[(size_t)n * ld + k0];
        b[2*r+1] = src[(size_t)n * ld + k0 + 1];
    }
    return b;
}

// Pre-swizzled weight: one contiguous 32B v16bf per lane per (nt,kt) tile.
static __device__ inline v16bf load_b_swz(const __bf16* wsw, int nt, int KT, int kk) {
    const v16bf* p = (const v16bf*)wsw;
    return p[((size_t)nt * KT + kk) * 32 + (threadIdx.x & 31)];
}

// ================= kernel 0: weight f32 -> bf16 B-fragment swizzle ==========
__global__ void k_swz(const float* __restrict__ w, __bf16* __restrict__ o,
                      int K, int N) {
    int KT = K >> 5;
    int tile = blockIdx.x;
    int nt = tile / KT, kt = tile - nt * KT;
    int tid = threadIdx.x;
    for (int idx = tid; idx < 512; idx += 256) {
        int lane = idx >> 4, e = idx & 15;
        int n = lane & 15, khalf = lane >> 4;
        int k = kt * 32 + khalf * 16 + e;
        o[(size_t)tile * 512 + idx] = (__bf16)w[(size_t)k * N + nt * 16 + n];
    }
}

// ================= kernel 1: null kv + zero padding =================
__global__ void k_init_kv(const float* __restrict__ null_kv,
                          __bf16* __restrict__ kb, __bf16* __restrict__ vb) {
    int bh = blockIdx.x;
    int d  = threadIdx.x;
    __bf16* kp = kb + (size_t)bh * CTXP * DH;
    __bf16* vp = vb + (size_t)bh * CTXP * DH;
    kp[d] = (__bf16)null_kv[d];
    vp[d] = (__bf16)null_kv[DH + d];
    for (int r = 1 + MM; r < CTXP; ++r) {
        kp[(size_t)r * DH + d] = (__bf16)0.0f;
        vp[(size_t)r * DH + d] = (__bf16)0.0f;
    }
}

// ================= kernel 2: kv = context @ W_kv =================
__global__ void __launch_bounds__(256)
k_kv_proj(const float* __restrict__ ctx, const __bf16* __restrict__ wkvs,
          __bf16* __restrict__ kb, __bf16* __restrict__ vb) {
    __shared__ __bf16 As[16][CTXD];
    const int KT = CTXD / 32;
    int b   = blockIdx.z;
    int m0  = blockIdx.y * 16;
    int cb  = blockIdx.x;
    int tid = threadIdx.x;
    for (int idx = tid; idx < 16 * CTXD; idx += 256) {
        int i = idx / CTXD, k = idx - i * CTXD;
        int m = m0 + i; if (m > MM - 1) m = MM - 1;
        As[i][k] = (__bf16)ctx[((size_t)b * MM + m) * CTXD + k];
    }
    __syncthreads();
    int wave = tid >> 5, lane = tid & 31;
    int c0  = cb * 512 + wave * 64;
    int nt0 = c0 >> 4;
    v8f acc[4];
#pragma unroll
    for (int t = 0; t < 4; ++t) acc[t] = {};
    for (int kk = 0; kk < KT; ++kk) {
        v16bf a = load_a_frag(&As[0][kk * 32], CTXD);
        if (kk + 1 < KT)
            __builtin_prefetch((const v16bf*)wkvs + ((size_t)nt0 * KT + kk + 1) * 32 + lane, 0, 0);
#pragma unroll
        for (int t = 0; t < 4; ++t) {
            v16bf bf = load_b_swz(wkvs, nt0 + t, KT, kk);
            acc[t] = wmma_bf16(a, bf, acc[t]);
        }
    }
    int half = lane >> 4, l15 = lane & 15;
#pragma unroll
    for (int t = 0; t < 4; ++t)
#pragma unroll
        for (int r = 0; r < 8; ++r) {
            int m = m0 + r + 8 * half;
            if (m >= MM) continue;
            int cc = c0 + t * 16 + l15;
            __bf16 v = (__bf16)acc[t][r];
            if (cc < INNER) {
                int h = cc >> 6, d = cc & 63;
                kb[(((size_t)b * NHEADS + h) * CTXP + 1 + m) * DH + d] = v;
            } else {
                int c2 = cc - INNER; int h = c2 >> 6, d = c2 & 63;
                vb[(((size_t)b * NHEADS + h) * CTXP + 1 + m) * DH + d] = v;
            }
        }
}

// ================= kernel 3: LN(x^T) then q = xn @ W_q =================
__global__ void __launch_bounds__(256)
k_ln_q(const float* __restrict__ x, const float* __restrict__ gamma,
       const __bf16* __restrict__ wqs, __bf16* __restrict__ qb) {
    __shared__ __bf16 Xs[16][CC];
    __shared__ float psum[16][17], psq[16][17];
    __shared__ float mu_s[16], rs_s[16];
    const int KT = CC / 32;
    int b = blockIdx.y, n0 = blockIdx.x * 16;
    int tid = threadIdx.x;
    int i = tid & 15, j = tid >> 4;
    const float* xb = x + (size_t)b * CC * NN;
    float s = 0.f, sq = 0.f;
    for (int c = j; c < CC; c += 16) {
        float v = xb[(size_t)c * NN + n0 + i];
        s += v; sq += v * v;
    }
    psum[i][j] = s; psq[i][j] = sq;
    __syncthreads();
    if (tid < 16) {
        float ts = 0.f, tq = 0.f;
        for (int jj = 0; jj < 16; ++jj) { ts += psum[tid][jj]; tq += psq[tid][jj]; }
        float mu = ts / CC;
        float var = tq / CC - mu * mu;
        mu_s[tid] = mu; rs_s[tid] = rsqrtf(var + EPSV);
    }
    __syncthreads();
    float mu = mu_s[i], rs = rs_s[i];
    for (int c = j; c < CC; c += 16) {
        float v = xb[(size_t)c * NN + n0 + i];
        Xs[i][c] = (__bf16)((v - mu) * rs * gamma[c]);
    }
    __syncthreads();
    int wave = tid >> 5, lane = tid & 31;
    int c0 = wave * 64;
    int nt0 = c0 >> 4;
    v8f acc[4];
#pragma unroll
    for (int t = 0; t < 4; ++t) acc[t] = {};
    for (int kk = 0; kk < KT; ++kk) {
        v16bf a = load_a_frag(&Xs[0][kk * 32], CC);
        if (kk + 1 < KT)
            __builtin_prefetch((const v16bf*)wqs + ((size_t)nt0 * KT + kk + 1) * 32 + lane, 0, 0);
#pragma unroll
        for (int t = 0; t < 4; ++t) {
            v16bf bf = load_b_swz(wqs, nt0 + t, KT, kk);
            acc[t] = wmma_bf16(a, bf, acc[t]);
        }
    }
    int half = lane >> 4, l15 = lane & 15;
#pragma unroll
    for (int t = 0; t < 4; ++t)
#pragma unroll
        for (int r = 0; r < 8; ++r) {
            int n  = n0 + r + 8 * half;
            int cc = c0 + t * 16 + l15;
            int h = cc >> 6, d = cc & 63;
            qb[(((size_t)b * NHEADS + h) * NN + n) * DH + d] = (__bf16)acc[t][r];
        }
}

// ================= kernel 4: flash attention =================
// compute-only: K/V already staged in LDS by the caller
template <int TILES, bool MASK>
__device__ inline void attn_compute(const v16bf (&aq)[2],
                                    const __bf16* __restrict__ Kc,
                                    const __bf16* __restrict__ Vc,
                                    __bf16* __restrict__ ps,   // wave's [16][PLD]
                                    float (&mi)[8], float (&li)[8],
                                    v8f (&oacc)[4], float scale) {
    int lane = threadIdx.x & 31;
    int half = lane >> 4, l15 = lane & 15;
    v8f s[TILES];
#pragma unroll
    for (int t = 0; t < TILES; ++t) {
        s[t] = {};
#pragma unroll
        for (int kk = 0; kk < 2; ++kk) {
            v16bf bf = load_bt_frag(Kc + (size_t)(t * 16) * DH + kk * 32, DH);
            s[t] = wmma_bf16(aq[kk], bf, s[t]);
        }
    }
    float mx[8];
#pragma unroll
    for (int r = 0; r < 8; ++r) mx[r] = -3.0e38f;
#pragma unroll
    for (int t = 0; t < TILES; ++t) {
        int j = t * 16 + l15;
#pragma unroll
        for (int r = 0; r < 8; ++r) {
            float v = s[t][r] * scale;
            if (MASK && j >= 1 + MM) v = -3.0e38f;
            s[t][r] = v;
            mx[r] = fmaxf(mx[r], v);
        }
    }
#pragma unroll
    for (int m = 1; m < 16; m <<= 1)
#pragma unroll
        for (int r = 0; r < 8; ++r)
            mx[r] = fmaxf(mx[r], __shfl_xor(mx[r], m, 32));
    float corr[8], rsum[8];
#pragma unroll
    for (int r = 0; r < 8; ++r) {
        float mn = fmaxf(mi[r], mx[r]);
        corr[r] = __expf(mi[r] - mn);
        mi[r] = mn;
        rsum[r] = 0.f;
    }
#pragma unroll
    for (int t = 0; t < TILES; ++t)
#pragma unroll
        for (int r = 0; r < 8; ++r) {
            float p = __expf(s[t][r] - mi[r]);
            rsum[r] += p;
            ps[(size_t)(r + 8 * half) * PLD + t * 16 + l15] = (__bf16)p;
        }
#pragma unroll
    for (int m = 1; m < 16; m <<= 1)
#pragma unroll
        for (int r = 0; r < 8; ++r)
            rsum[r] += __shfl_xor(rsum[r], m, 32);
#pragma unroll
    for (int r = 0; r < 8; ++r) li[r] = li[r] * corr[r] + rsum[r];
#pragma unroll
    for (int t = 0; t < 4; ++t)
#pragma unroll
        for (int r = 0; r < 8; ++r) oacc[t][r] = oacc[t][r] * corr[r];
    __syncthreads();                      // P visible; uniform across waves
    constexpr int KST = (TILES * 16) / 32;
#pragma unroll
    for (int kk = 0; kk < KST; ++kk) {
        v16bf a2 = load_a_frag(ps + kk * 32, PLD);
#pragma unroll
        for (int t = 0; t < 4; ++t) {
            v16bf bf = load_b_frag(Vc + (size_t)(kk * 32) * DH + t * 16, DH);
            oacc[t] = wmma_bf16(a2, bf, oacc[t]);
        }
    }
    __syncthreads();                      // all waves done with this K/V buffer
}

// grid (N/64, NHEADS, B), 128 thr (4 waves x 16 query rows)
__global__ void __launch_bounds__(128)
k_attn(const __bf16* __restrict__ qb, const __bf16* __restrict__ kb,
       const __bf16* __restrict__ vb, __bf16* __restrict__ ao) {
    __shared__ __bf16 Ps[4][16 * PLD];
    __shared__ __bf16 Kc[CTXP * DH];
    __shared__ __bf16 Vc[CTXP * DH];
    __shared__ __bf16 Kq[2][64 * DH];     // double buffer for self-key chunks
    int b = blockIdx.z, h = blockIdx.y, q0 = blockIdx.x * 64;
    int tid = threadIdx.x, wave = tid >> 5, lane = tid & 31;
    const __bf16* qh = qb + ((size_t)b * NHEADS + h) * NN * DH;
    const __bf16* kh = kb + ((size_t)b * NHEADS + h) * CTXP * DH;
    const __bf16* vh = vb + ((size_t)b * NHEADS + h) * CTXP * DH;
    int qr0 = q0 + wave * 16;
    v16bf aq[2];
    aq[0] = load_a_frag(qh + (size_t)qr0 * DH, DH);
    aq[1] = load_a_frag(qh + (size_t)qr0 * DH + 32, DH);
    float mi[8], li[8];
    v8f oacc[4];
#pragma unroll
    for (int r = 0; r < 8; ++r) { mi[r] = -3.0e38f; li[r] = 0.f; }
#pragma unroll
    for (int t = 0; t < 4; ++t) oacc[t] = {};
    const float scale = 0.125f;
    __bf16* ps = &Ps[wave][0];

    // phase 1: null + context keys (96 padded rows, masked)
    if (wave == 0) {
        tdm_load_1d(kh, Kc, CTXP * DH * sizeof(__bf16));
        tdm_load_1d(vh, Vc, CTXP * DH * sizeof(__bf16));
        tdm_wait();
    }
    __syncthreads();
    attn_compute<6, true>(aq, Kc, Vc, ps, mi, li, oacc, scale);

    // phase 2: queries as keys/values, TDM double-buffered
    if (wave == 0) tdm_load_1d(qh, &Kq[0][0], 64 * DH * sizeof(__bf16));
    for (int c = 0; c < NN / 64; ++c) {
        if (wave == 0) tdm_wait();        // chunk c staged
        __syncthreads();
        if (wave == 0 && c + 1 < NN / 64) // overlap next DMA with compute
            tdm_load_1d(qh + (size_t)((c + 1) * 64) * DH,
                        &Kq[(c + 1) & 1][0], 64 * DH * sizeof(__bf16));
        const __bf16* kc = &Kq[c & 1][0];
        attn_compute<4, false>(aq, kc, kc, ps, mi, li, oacc, scale);
    }

    int half = lane >> 4, l15 = lane & 15;
    float linv[8];
#pragma unroll
    for (int r = 0; r < 8; ++r) linv[r] = 1.0f / li[r];
#pragma unroll
    for (int t = 0; t < 4; ++t)
#pragma unroll
        for (int r = 0; r < 8; ++r) {
            int n   = qr0 + r + 8 * half;
            int col = h * DH + t * 16 + l15;
            ao[((size_t)b * NN + n) * INNER + col] = (__bf16)(oacc[t][r] * linv[r]);
        }
}

// ========= kernel 5: out@W_out, LN, residual, transposed store =========
__global__ void __launch_bounds__(256)
k_out(const __bf16* __restrict__ ao, const __bf16* __restrict__ wos,
      const float* __restrict__ gout, const float* __restrict__ x,
      float* __restrict__ y) {
    __shared__ __bf16 As[16][INNER];
    __shared__ float rsum[16], rsq[16];
    __shared__ float mu_s[16], rs_s[16];
    const int KT = INNER / 32;
    int b = blockIdx.y, n0 = blockIdx.x * 16;
    int tid = threadIdx.x;
    if (tid < 32) {
        tdm_load_1d(ao + ((size_t)b * NN + n0) * INNER, &As[0][0],
                    16 * INNER * sizeof(__bf16));
        tdm_wait();
    }
    if (tid < 16) { rsum[tid] = 0.f; rsq[tid] = 0.f; }
    __syncthreads();
    int wave = tid >> 5, lane = tid & 31;
    int half = lane >> 4, l15 = lane & 15;
    int c0 = wave * 64;
    int nt0 = c0 >> 4;
    v8f acc[4];
#pragma unroll
    for (int t = 0; t < 4; ++t) acc[t] = {};
    for (int kk = 0; kk < KT; ++kk) {
        v16bf a = load_a_frag(&As[0][kk * 32], INNER);
        if (kk + 1 < KT)
            __builtin_prefetch((const v16bf*)wos + ((size_t)nt0 * KT + kk + 1) * 32 + lane, 0, 0);
#pragma unroll
        for (int t = 0; t < 4; ++t) {
            v16bf bf = load_b_swz(wos, nt0 + t, KT, kk);
            acc[t] = wmma_bf16(a, bf, acc[t]);
        }
    }
    float s[8], sq[8];
#pragma unroll
    for (int r = 0; r < 8; ++r) { s[r] = 0.f; sq[r] = 0.f; }
#pragma unroll
    for (int t = 0; t < 4; ++t)
#pragma unroll
        for (int r = 0; r < 8; ++r) { float v = acc[t][r]; s[r] += v; sq[r] += v * v; }
#pragma unroll
    for (int m = 1; m < 16; m <<= 1)
#pragma unroll
        for (int r = 0; r < 8; ++r) {
            s[r]  += __shfl_xor(s[r],  m, 32);
            sq[r] += __shfl_xor(sq[r], m, 32);
        }
    if (l15 == 0)
#pragma unroll
        for (int r = 0; r < 8; ++r) {
            atomicAdd(&rsum[r + 8 * half], s[r]);
            atomicAdd(&rsq[r + 8 * half], sq[r]);
        }
    __syncthreads();
    if (tid < 16) {
        float mu  = rsum[tid] / CC;
        float var = rsq[tid] / CC - mu * mu;
        mu_s[tid] = mu; rs_s[tid] = rsqrtf(var + EPSV);
    }
    __syncthreads();
    const float* xb = x + (size_t)b * CC * NN;
    float* yb = y + (size_t)b * CC * NN;
#pragma unroll
    for (int t = 0; t < 4; ++t)
#pragma unroll
        for (int r = 0; r < 8; ++r) {
            int row = r + 8 * half;
            int col = c0 + t * 16 + l15;
            float o = (acc[t][r] - mu_s[row]) * rs_s[row] * gout[col];
            yb[(size_t)col * NN + n0 + row] = xb[(size_t)col * NN + n0 + row] + o;
        }
}

// ================= launcher =================
extern "C" void kernel_launch(void* const* d_in, const int* in_sizes, int n_in,
                              void* d_out, int out_size, void* d_ws, size_t ws_size,
                              hipStream_t stream) {
    const float* x     = (const float*)d_in[0];
    const float* ctx   = (const float*)d_in[1];
    const float* gnorm = (const float*)d_in[2];
    const float* nkv   = (const float*)d_in[3];
    const float* wq    = (const float*)d_in[4];
    const float* wkv   = (const float*)d_in[5];
    const float* wout  = (const float*)d_in[6];
    const float* gout  = (const float*)d_in[7];
    float* out = (float*)d_out;

    char* ws = (char*)d_ws;
    size_t off = 0;
    __bf16* qb   = (__bf16*)(ws + off); off += (size_t)BB * NHEADS * NN * DH * sizeof(__bf16);
    __bf16* kb   = (__bf16*)(ws + off); off += (size_t)BB * NHEADS * CTXP * DH * sizeof(__bf16);
    __bf16* vb   = (__bf16*)(ws + off); off += (size_t)BB * NHEADS * CTXP * DH * sizeof(__bf16);
    __bf16* ao   = (__bf16*)(ws + off); off += (size_t)BB * NN * INNER * sizeof(__bf16);
    __bf16* wqs  = (__bf16*)(ws + off); off += (size_t)CC * INNER * sizeof(__bf16);
    __bf16* wkvs = (__bf16*)(ws + off); off += (size_t)CTXD * 1024 * sizeof(__bf16);
    __bf16* wos  = (__bf16*)(ws + off); off += (size_t)INNER * CC * sizeof(__bf16);

    k_swz<<<(CC / 32) * (INNER / 16), 256, 0, stream>>>(wq, wqs, CC, INNER);
    k_swz<<<(CTXD / 32) * (1024 / 16), 256, 0, stream>>>(wkv, wkvs, CTXD, 1024);
    k_swz<<<(INNER / 32) * (CC / 16), 256, 0, stream>>>(wout, wos, INNER, CC);

    k_init_kv<<<BB * NHEADS, DH, 0, stream>>>(nkv, kb, vb);
    k_kv_proj<<<dim3(2, 5, BB), 256, 0, stream>>>(ctx, wkvs, kb, vb);
    k_ln_q<<<dim3(NN / 16, BB), 256, 0, stream>>>(x, gnorm, wqs, qb);
    k_attn<<<dim3(NN / 64, NHEADS, BB), 128, 0, stream>>>(qb, kb, vb, ao);
    k_out<<<dim3(NN / 16, BB), 256, 0, stream>>>(ao, wos, gout, x, out);
}
